// DIEN_55130200211740
// MI455X (gfx1250) — matrix operand order repaired and checked by reading, not compiled, not run on hardware
//
#include <hip/hip_runtime.h>
#include <hip/hip_bf16.h>
#include <cstdint>

// ---------------- CDNA5 WMMA types ----------------
typedef __bf16  bf16_t;
typedef bf16_t  v16bf __attribute__((ext_vector_type(16)));
typedef float   v8f   __attribute__((ext_vector_type(8)));
typedef int     v4i   __attribute__((ext_vector_type(4)));

#define AS1 __attribute__((address_space(1)))
#define AS3 __attribute__((address_space(3)))

#if __has_builtin(__builtin_amdgcn_global_load_async_to_lds_b128)
#define HAVE_ASYNC_LDS 1
#else
#define HAVE_ASYNC_LDS 0
#endif

__device__ __forceinline__ void async_b128(const unsigned short* g, unsigned short* l) {
#if HAVE_ASYNC_LDS
    __builtin_amdgcn_global_load_async_to_lds_b128(
        (AS1 v4i*)(unsigned short*)g, (AS3 v4i*)l, 0, 0);
#endif
}
__device__ __forceinline__ void wait_async0() {
#if HAVE_ASYNC_LDS
#if __has_builtin(__builtin_amdgcn_s_wait_asynccnt)
    __builtin_amdgcn_s_wait_asynccnt(0);
#else
    asm volatile("s_wait_asynccnt 0x0" ::: "memory");
#endif
#endif
}

// Problem dims (fixed by the reference)
constexpr int Bsz = 512;
constexpr int T   = 256;
constexpr int D   = 256;
constexpr int H   = 256;
constexpr int G   = 3 * H;   // 768 gates
constexpr int KK  = 512;     // fused K = D + H

// scan tiling
constexpr int RB   = 32;     // batch rows per block -> 16 blocks
constexpr int AKS  = 264;    // A-operand LDS row stride (shorts)
constexpr int BNS  = 40;     // B LDS per-column stride (shorts); conflict-free b128
constexpr int BTS  = 16 * BNS;          // one 32x16 B tile = 640 shorts
constexpr int BBUF = 48 * BTS;          // 48 tiles per K-step = 30720 shorts (60 KB)

__device__ __forceinline__ unsigned f2bf1(float x) {
    unsigned u = __float_as_uint(x);
    return (u + 0x7FFFu + ((u >> 16) & 1u)) >> 16;   // RNE
}
__device__ __forceinline__ unsigned pk2(float a, float b) {
    return f2bf1(a) | (f2bf1(b) << 16);
}
__device__ __forceinline__ float sigf(float x) {
    return 1.0f / (1.0f + __expf(-x));
}

// ---------------- kernel 1: wu_bf16 = [w | u] (G x 512) ----------------
__global__ void convert_wu_kernel(const float* __restrict__ w,
                                  const float* __restrict__ u,
                                  unsigned short* __restrict__ wu) {
    int idx = blockIdx.x * 256 + threadIdx.x;      // G*KK = 393216 total
    int g = idx >> 9;          // / 512
    int k = idx & 511;
    float v = (k < D) ? w[g * D + k] : u[g * H + (k - D)];
    wu[idx] = (unsigned short)f2bf1(v);
}

// ---------------- kernel 2: q_proj = query @ w_att^T ----------------
__global__ __launch_bounds__(256) void qproj_kernel(const float* __restrict__ query,
                                                    const float* __restrict__ w_att,
                                                    float* __restrict__ qp) {
    __shared__ float qrow[D];
    int b = blockIdx.x, tid = threadIdx.x;
    qrow[tid] = query[b * D + tid];
    __syncthreads();
    const float* wr = w_att + (size_t)tid * D;
    float acc = 0.0f;
#pragma unroll 4
    for (int k = 0; k < D; ++k) acc += qrow[k] * wr[k];
    qp[b * D + tid] = acc;
}

// ---------------- kernel 3: logits + masked softmax (exact zeros) ----------------
__global__ __launch_bounds__(256) void attn_kernel(const float* __restrict__ qp,
                                                   const float* __restrict__ keys,
                                                   const int* __restrict__ seq_len,
                                                   float* __restrict__ score) {
    __shared__ float q_s[D];
    __shared__ float lg[T];
    __shared__ float red[8];
    int b = blockIdx.x, tid = threadIdx.x;
    int lane = tid & 31, wave = tid >> 5;

    q_s[tid] = qp[b * D + tid];
    __syncthreads();

    const float* kb = keys + (size_t)b * T * D;
    for (int t = wave; t < T; t += 8) {
        const float* kr = kb + (size_t)t * D;
        float acc = 0.0f;
        for (int k = lane; k < D; k += 32) acc += q_s[k] * kr[k];
        for (int o = 16; o; o >>= 1) acc += __shfl_down(acc, o, 32);
        if (lane == 0) lg[t] = acc;
    }
    __syncthreads();

    int L = seq_len[b];
    float v = (tid < L) ? lg[tid] : -3.0e38f;
    float m = v;
    for (int o = 16; o; o >>= 1) m = fmaxf(m, __shfl_down(m, o, 32));
    if (lane == 0) red[wave] = m;
    __syncthreads();
    m = red[0];
#pragma unroll
    for (int i = 1; i < 8; ++i) m = fmaxf(m, red[i]);
    __syncthreads();

    float e = (tid < L) ? __expf(v - m) : 0.0f;   // exact 0 when masked
    float s = e;
    for (int o = 16; o; o >>= 1) s += __shfl_down(s, o, 32);
    if (lane == 0) red[wave] = s;
    __syncthreads();
    s = 0.0f;
#pragma unroll
    for (int i = 0; i < 8; ++i) s += red[i];

    score[(size_t)b * T + tid] = e / s;
}

// ---------------- kernel 4: persistent fused AUGRU scan ----------------
// grid = 16 blocks x 256 threads (8 waves). Block owns 32 batch rows for all T steps.
// Per step: gates = [keys_t | h](32x512) @ wu^T(512x768), bf16 WMMA, f32 accum.
// B (wu) K-slices staged global->LDS with GLOBAL_LOAD_ASYNC_TO_LDS_B128 (ASYNCcnt),
// double buffered, rotated seamlessly across t. z-gate K-halves kept separate so the
// reset gate r can scale h_z before tanh. h is register-resident per owning lane.
__global__ __launch_bounds__(256) void scan_kernel(const float* __restrict__ keys,
                                                   const float* __restrict__ score,
                                                   const unsigned short* __restrict__ wu,
                                                   const float* __restrict__ bu,
                                                   const float* __restrict__ br,
                                                   const float* __restrict__ bh,
                                                   float* __restrict__ out) {
    __shared__ unsigned short kq[2][RB * AKS];   // keys_t bf16, double buffered (A, K<256)
    __shared__ unsigned short hbuf[RB * AKS];    // h bf16 (A, K>=256)
    __shared__ unsigned short bb[2][BBUF];       // wu K-slice, double buffered (B)
    __shared__ float s_lds[RB];

    const int tid   = threadIdx.x;
    const int lane  = tid & 31;
    const int wave  = tid >> 5;
    const int b0    = blockIdx.x * RB;

    const int g0    = 2 * wave;           // gate-triple groups g0, g0+1
    const int n15   = lane & 15;
    const int khalf = lane >> 4;

    // keys staging: thread -> (row, 32-float segment)
    const int krow = tid >> 3;            // 0..31
    const int kcol = (tid & 7) * 32;      // 0,32,...,224

    // cooperative B staging: thread -> (chunk c, column n, tile set ct0+4j)
    const int cc  = tid & 3;              // 16B chunk within 64B column row
    const int cn  = (tid >> 2) & 15;      // B column
    const int ct0 = tid >> 6;             // 0..3 ; tiles ct0 + 4*j, j=0..11

    // biases for owned gate columns
    float bus[2], brs[2], bhs[2];
#pragma unroll
    for (int i = 0; i < 2; ++i) {
        int j = 16 * (g0 + i) + n15;
        bus[i] = bu[j]; brs[i] = br[j]; bhs[i] = bh[j];
    }

    // register-resident h: hreg[i][m][v] = h[16m + v + 8*khalf][16(g0+i)+n15]
    float hreg[2][2][8];
#pragma unroll
    for (int i = 0; i < 2; ++i)
#pragma unroll
        for (int m = 0; m < 2; ++m)
#pragma unroll
            for (int v = 0; v < 8; ++v) hreg[i][m][v] = 0.0f;

    // ---- init: zero h buffer, stage keys[0], prime B buffer 0 with kb=0 ----
    for (int idx = tid; idx < RB * AKS; idx += 256) hbuf[idx] = 0;
    {
        const uint4* kp = (const uint4*)(keys + ((size_t)(b0 + krow) * T + 0) * D + kcol);
#pragma unroll
        for (int j = 0; j < 4; ++j) {
            uint4 a = kp[2 * j], b = kp[2 * j + 1];
            uint4 o;
            o.x = pk2(__uint_as_float(a.x), __uint_as_float(a.y));
            o.y = pk2(__uint_as_float(a.z), __uint_as_float(a.w));
            o.z = pk2(__uint_as_float(b.x), __uint_as_float(b.y));
            o.w = pk2(__uint_as_float(b.z), __uint_as_float(b.w));
            *(uint4*)(&kq[0][krow * AKS + kcol + 8 * j]) = o;
        }
    }
#pragma unroll 1
    for (int j = 0; j < 12; ++j) {
        int ct = ct0 + 4 * j;
        const unsigned short* src = wu + (size_t)(16 * ct + cn) * KK + 0 + cc * 8;
        unsigned short* dst = &bb[0][ct * BTS + cn * BNS + cc * 8];
#if HAVE_ASYNC_LDS
        async_b128(src, dst);
#else
        *(uint4*)dst = *(const uint4*)src;
#endif
    }
    wait_async0();
    __syncthreads();

    const v8f vzero = {0.f, 0.f, 0.f, 0.f, 0.f, 0.f, 0.f, 0.f};

    for (int t = 0; t < T; ++t) {
        asm volatile("" ::: "memory");   // stop LICM from hoisting/spilling B loads across t

        // score for this step (read after the K-loop barriers)
        if (tid < RB) s_lds[tid] = score[(size_t)(b0 + tid) * T + t];

        // prefetch keys[t+1] into registers (hidden under the GEMM)
        int tn = (t + 1 < T) ? (t + 1) : (T - 1);
        uint4 pf[8];
        {
            const uint4* kp = (const uint4*)(keys + ((size_t)(b0 + krow) * T + tn) * D + kcol);
#pragma unroll
            for (int j = 0; j < 8; ++j) pf[j] = kp[j];
        }

        v8f au[2][2], ar[2][2], azi[2][2], azh[2][2];
#pragma unroll
        for (int i = 0; i < 2; ++i)
#pragma unroll
            for (int m = 0; m < 2; ++m) { au[i][m] = vzero; ar[i][m] = vzero;
                                          azi[i][m] = vzero; azh[i][m] = vzero; }

        const unsigned short* kqb = kq[t & 1];

        // ---------- K-half 1: A = keys_t (K 0..255), z -> azi ----------
#pragma unroll 1
        for (int kb16 = 0; kb16 < 8; ++kb16) {
            const int p = kb16 & 1, pn = p ^ 1;
            const int kb = kb16 * 32;
            const int kbn = kb + 32;                 // next slice (256 at kb16=7 -> half 2)
#if HAVE_ASYNC_LDS
#pragma unroll
            for (int j = 0; j < 12; ++j) {
                int ct = ct0 + 4 * j;
                async_b128(wu + (size_t)(16 * ct + cn) * KK + kbn + cc * 8,
                           &bb[pn][ct * BTS + cn * BNS + cc * 8]);
            }
#else
            uint4 breg[12];
#pragma unroll
            for (int j = 0; j < 12; ++j) {
                int ct = ct0 + 4 * j;
                breg[j] = *(const uint4*)(wu + (size_t)(16 * ct + cn) * KK + kbn + cc * 8);
            }
#endif
            union { v16bf v; uint4 q[2]; } fa0, fa1, fb;
            fa0.q[0] = *(const uint4*)(kqb + n15 * AKS + kb + 8 * khalf);
            fa0.q[1] = *(const uint4*)(kqb + n15 * AKS + kb + 16 + 8 * khalf);
            fa1.q[0] = *(const uint4*)(kqb + (16 + n15) * AKS + kb + 8 * khalf);
            fa1.q[1] = *(const uint4*)(kqb + (16 + n15) * AKS + kb + 16 + 8 * khalf);
#pragma unroll
            for (int i = 0; i < 2; ++i) {
                const int g = g0 + i;
                int off = g * BTS + n15 * BNS + khalf * 8;
                fb.q[0] = *(const uint4*)(&bb[p][off]);
                fb.q[1] = *(const uint4*)(&bb[p][off + 16]);
                au[i][0] = __builtin_amdgcn_wmma_f32_16x16x32_bf16(false, fa0.v, false, fb.v, (short)0, au[i][0], false, false);
                au[i][1] = __builtin_amdgcn_wmma_f32_16x16x32_bf16(false, fa1.v, false, fb.v, (short)0, au[i][1], false, false);
                off = (16 + g) * BTS + n15 * BNS + khalf * 8;
                fb.q[0] = *(const uint4*)(&bb[p][off]);
                fb.q[1] = *(const uint4*)(&bb[p][off + 16]);
                ar[i][0] = __builtin_amdgcn_wmma_f32_16x16x32_bf16(false, fa0.v, false, fb.v, (short)0, ar[i][0], false, false);
                ar[i][1] = __builtin_amdgcn_wmma_f32_16x16x32_bf16(false, fa1.v, false, fb.v, (short)0, ar[i][1], false, false);
                off = (32 + g) * BTS + n15 * BNS + khalf * 8;
                fb.q[0] = *(const uint4*)(&bb[p][off]);
                fb.q[1] = *(const uint4*)(&bb[p][off + 16]);
                azi[i][0] = __builtin_amdgcn_wmma_f32_16x16x32_bf16(false, fa0.v, false, fb.v, (short)0, azi[i][0], false, false);
                azi[i][1] = __builtin_amdgcn_wmma_f32_16x16x32_bf16(false, fa1.v, false, fb.v, (short)0, azi[i][1], false, false);
            }
#if HAVE_ASYNC_LDS
            wait_async0();
#else
#pragma unroll
            for (int j = 0; j < 12; ++j) {
                int ct = ct0 + 4 * j;
                *(uint4*)(&bb[pn][ct * BTS + cn * BNS + cc * 8]) = breg[j];
            }
#endif
            __syncthreads();
        }

        // ---------- K-half 2: A = h (K 256..511), z -> azh ----------
#pragma unroll 1
        for (int kb16 = 8; kb16 < 16; ++kb16) {
            const int p = kb16 & 1, pn = p ^ 1;
            const int kb = kb16 * 32;
            const int ko = kb - 256;
            const int kbn = (kb + 32) & 511;         // wraps to 0 at the last slice
#if HAVE_ASYNC_LDS
#pragma unroll
            for (int j = 0; j < 12; ++j) {
                int ct = ct0 + 4 * j;
                async_b128(wu + (size_t)(16 * ct + cn) * KK + kbn + cc * 8,
                           &bb[pn][ct * BTS + cn * BNS + cc * 8]);
            }
#else
            uint4 breg[12];
#pragma unroll
            for (int j = 0; j < 12; ++j) {
                int ct = ct0 + 4 * j;
                breg[j] = *(const uint4*)(wu + (size_t)(16 * ct + cn) * KK + kbn + cc * 8);
            }
#endif
            union { v16bf v; uint4 q[2]; } fa0, fa1, fb;
            fa0.q[0] = *(const uint4*)(hbuf + n15 * AKS + ko + 8 * khalf);
            fa0.q[1] = *(const uint4*)(hbuf + n15 * AKS + ko + 16 + 8 * khalf);
            fa1.q[0] = *(const uint4*)(hbuf + (16 + n15) * AKS + ko + 8 * khalf);
            fa1.q[1] = *(const uint4*)(hbuf + (16 + n15) * AKS + ko + 16 + 8 * khalf);
#pragma unroll
            for (int i = 0; i < 2; ++i) {
                const int g = g0 + i;
                int off = g * BTS + n15 * BNS + khalf * 8;
                fb.q[0] = *(const uint4*)(&bb[p][off]);
                fb.q[1] = *(const uint4*)(&bb[p][off + 16]);
                au[i][0] = __builtin_amdgcn_wmma_f32_16x16x32_bf16(false, fa0.v, false, fb.v, (short)0, au[i][0], false, false);
                au[i][1] = __builtin_amdgcn_wmma_f32_16x16x32_bf16(false, fa1.v, false, fb.v, (short)0, au[i][1], false, false);
                off = (16 + g) * BTS + n15 * BNS + khalf * 8;
                fb.q[0] = *(const uint4*)(&bb[p][off]);
                fb.q[1] = *(const uint4*)(&bb[p][off + 16]);
                ar[i][0] = __builtin_amdgcn_wmma_f32_16x16x32_bf16(false, fa0.v, false, fb.v, (short)0, ar[i][0], false, false);
                ar[i][1] = __builtin_amdgcn_wmma_f32_16x16x32_bf16(false, fa1.v, false, fb.v, (short)0, ar[i][1], false, false);
                off = (32 + g) * BTS + n15 * BNS + khalf * 8;
                fb.q[0] = *(const uint4*)(&bb[p][off]);
                fb.q[1] = *(const uint4*)(&bb[p][off + 16]);
                azh[i][0] = __builtin_amdgcn_wmma_f32_16x16x32_bf16(false, fa0.v, false, fb.v, (short)0, azh[i][0], false, false);
                azh[i][1] = __builtin_amdgcn_wmma_f32_16x16x32_bf16(false, fa1.v, false, fb.v, (short)0, azh[i][1], false, false);
            }
#if HAVE_ASYNC_LDS
            wait_async0();
#else
#pragma unroll
            for (int j = 0; j < 12; ++j) {
                int ct = ct0 + 4 * j;
                *(uint4*)(&bb[pn][ct * BTS + cn * BNS + cc * 8]) = breg[j];
            }
#endif
            __syncthreads();
        }

        // ---------- gates + h update (register-resident h) ----------
#pragma unroll
        for (int i = 0; i < 2; ++i) {
            const int c = 16 * (g0 + i) + n15;
#pragma unroll
            for (int m = 0; m < 2; ++m)
#pragma unroll
                for (int v = 0; v < 8; ++v) {
                    int r_loc = 16 * m + v + 8 * khalf;
                    float sv   = s_lds[r_loc];               // exact 0 when masked
                    float hold = hreg[i][m][v];
                    float ug   = sv * sigf(au[i][m][v] + bus[i]);
                    float rr   = sigf(ar[i][m][v] + brs[i]);
                    float zz   = tanhf(azi[i][m][v] + rr * azh[i][m][v] + bhs[i]);
                    float hn   = hold + ug * (zz - hold);
                    hreg[i][m][v] = hn;
                    hbuf[r_loc * AKS + c] = (unsigned short)f2bf1(hn);
                }
        }
        // store prefetched keys[t+1] into the other keys buffer
        {
            unsigned short* kdst = kq[(t + 1) & 1];
#pragma unroll
            for (int j = 0; j < 4; ++j) {
                uint4 a = pf[2 * j], b = pf[2 * j + 1];
                uint4 o;
                o.x = pk2(__uint_as_float(a.x), __uint_as_float(a.y));
                o.y = pk2(__uint_as_float(a.z), __uint_as_float(a.w));
                o.z = pk2(__uint_as_float(b.x), __uint_as_float(b.y));
                o.w = pk2(__uint_as_float(b.z), __uint_as_float(b.w));
                *(uint4*)(&kdst[krow * AKS + kcol + 8 * j]) = o;
            }
        }
        __syncthreads();   // h + keys writes visible before next step's GEMM
    }

    // out = final h (== h at t = seq_len-1, frozen afterwards)
#pragma unroll
    for (int i = 0; i < 2; ++i) {
        const int c = 16 * (g0 + i) + n15;
#pragma unroll
        for (int m = 0; m < 2; ++m)
#pragma unroll
            for (int v = 0; v < 8; ++v) {
                int r = 16 * m + v + 8 * khalf;
                out[(size_t)(b0 + r) * H + c] = hreg[i][m][v];
            }
    }
}

// ---------------- launch ----------------
extern "C" void kernel_launch(void* const* d_in, const int* in_sizes, int n_in,
                              void* d_out, int out_size, void* d_ws, size_t ws_size,
                              hipStream_t stream) {
    (void)in_sizes; (void)n_in; (void)out_size; (void)ws_size;
    const float* query   = (const float*)d_in[0];
    const float* keys    = (const float*)d_in[1];
    const int*   seq_len = (const int*)d_in[2];
    const float* w_att   = (const float*)d_in[3];
    const float* w       = (const float*)d_in[4];
    const float* u       = (const float*)d_in[5];
    const float* bu      = (const float*)d_in[6];
    const float* br      = (const float*)d_in[7];
    const float* bh      = (const float*)d_in[8];
    float* out = (float*)d_out;

    float* qp    = (float*)d_ws;                         // 512*256 f32
    float* score = qp + (size_t)Bsz * D;                 // 512*256 f32
    unsigned short* wu = (unsigned short*)(score + (size_t)Bsz * T);  // 768*512 bf16

    convert_wu_kernel<<<(G * KK) / 256, 256, 0, stream>>>(w, u, wu);
    qproj_kernel<<<Bsz, 256, 0, stream>>>(query, w_att, qp);
    attn_kernel<<<Bsz, 256, 0, stream>>>(qp, keys, seq_len, score);
    scan_kernel<<<Bsz / RB, 256, 0, stream>>>(keys, score, wu, bu, br, bh, out);
}
